// TGAT_4440996184764
// MI455X (gfx1250) — compile-verified, matrix operand order checked by reading, not run on hardware
//
#include <hip/hip_runtime.h>
#include <hip/hip_bf16.h>
#include <math.h>

typedef __bf16 bf16;
typedef __attribute__((ext_vector_type(16))) bf16  v16bf;
typedef __attribute__((ext_vector_type(8)))  float v8f;

#define B_SZ   4096
#define K_NBR  10
#define ND_    128
#define ED_    128
#define TD_    100
#define EMB_   128
#define QD_    228          // ND + TD
#define KD_    356          // ND + ED + TD
#define DH_    114          // QD / 2 heads

// ---------------------------------------------------------------------------
// A-matrix segment descriptor: A rows are assembled from up to two column
// segments (direct rows, row-gathered rows, or a broadcast row vector).
// ---------------------------------------------------------------------------
struct Seg {
  const float* ptr;
  const int*   gidx;   // non-null -> row index = gidx[r]
  int          cols;
  int          ld;
  int          bcast;  // 1 -> A[r,c] = ptr[c] for all rows
};

static Seg segG(const float* p, const int* g, int cols, int ld) {
  Seg s; s.ptr = p; s.gidx = g; s.cols = cols; s.ld = ld; s.bcast = 0; return s;
}
static Seg segD(const float* p, int cols, int ld) { return segG(p, nullptr, cols, ld); }
static Seg segB(const float* p, int cols) {
  Seg s; s.ptr = p; s.gidx = nullptr; s.cols = cols; s.ld = 0; s.bcast = 1; return s;
}
static Seg segN() { Seg s; s.ptr = nullptr; s.gidx = nullptr; s.cols = 0; s.ld = 0; s.bcast = 0; return s; }

// ---------------------------------------------------------------------------
// pack_a: assemble A (segments / gathers / broadcast) once into a padded
// row-major bf16 buffer Ap[Mpad][Kpad]. All branchy work lives here,
// touched exactly once per element. Rows >= M and cols >= Kt are zeroed.
// ---------------------------------------------------------------------------
__global__ __launch_bounds__(256) void pack_a_kernel(
    Seg s0, Seg s1, int M, int Kt, int Kpad, long Mpad, bf16* __restrict__ Ap)
{
  const long idx = (long)blockIdx.x * 256 + threadIdx.x;
  const int  groups = Kpad >> 3;
  const long row = idx / groups;
  const int  kg  = (int)(idx % groups);
  if (row >= Mpad) return;

  union { unsigned short u[8]; uint4 q; } o;
  if (row < M) {
    long roff0 = 0, roff1 = 0;
    if (s0.cols > 0 && !s0.bcast)
      roff0 = (long)(s0.gidx ? s0.gidx[row] : (int)row) * s0.ld;
    if (s1.cols > 0 && !s1.bcast)
      roff1 = (long)(s1.gidx ? s1.gidx[row] : (int)row) * s1.ld;
    #pragma unroll
    for (int i = 0; i < 8; ++i) {
      const int k = kg * 8 + i;
      float v = 0.f;
      if (k < Kt) {
        if (k < s0.cols) {
          v = s0.bcast ? s0.ptr[k] : s0.ptr[roff0 + k];
        } else {
          const int lc = k - s0.cols;
          if (lc < s1.cols) v = s1.bcast ? s1.ptr[lc] : s1.ptr[roff1 + lc];
        }
      }
      const bf16 b = (bf16)v;
      o.u[i] = __builtin_bit_cast(unsigned short, b);
    }
  } else {
    o.q = uint4{0u, 0u, 0u, 0u};
  }
  *(uint4*)&Ap[row * Kpad + (long)kg * 8] = o.q;
}

// ---------------------------------------------------------------------------
// pack_w: W[Kt x N] (optionally given transposed) -> bf16 Wt[Npad][Kpad]
// (n-major, k-contiguous: exactly the B-fragment feed layout).
// ---------------------------------------------------------------------------
__global__ __launch_bounds__(256) void pack_w_kernel(
    const float* __restrict__ W, int ldw, int wtrans, int Kt, int N,
    int Kpad, int Npad, bf16* __restrict__ Wt)
{
  const int idx = blockIdx.x * 256 + threadIdx.x;
  const int groups = Kpad >> 3;
  const int row = idx / groups;           // n index
  const int kg  = idx % groups;
  if (row >= Npad) return;

  union { unsigned short u[8]; uint4 q; } o;
  #pragma unroll
  for (int i = 0; i < 8; ++i) {
    const int k = kg * 8 + i;
    float v = 0.f;
    if (row < N && k < Kt)
      v = wtrans ? W[(long)row * ldw + k] : W[(long)k * ldw + row];
    const bf16 b = (bf16)v;
    o.u[i] = __builtin_bit_cast(unsigned short, b);
  }
  *(uint4*)&Wt[(long)row * Kpad + (long)kg * 8] = o.q;
}

// ---------------------------------------------------------------------------
// Branch-free WMMA GEMM on packed bf16 operands.
// Block = 256 threads (8 waves); block tile 128(M) x 64(N); each wave owns a
// 32x32 subtile = 4 accumulators with 2x2 fragment reuse. Steady-state loop:
// 8x global_load_b128 + 4x v_wmma_f32_16x16x32_bf16. Operands are padded so
// no bounds checks are needed on loads; only the fp32 epilogue stores check.
// ---------------------------------------------------------------------------
__global__ __launch_bounds__(256) void gemm_wmma_packed(
    const bf16* __restrict__ A, const bf16* __restrict__ Wt,
    int M, int N, int Kpad,
    const float* __restrict__ bias, float scale, int relu,
    const int* __restrict__ rowmask, float* __restrict__ C, int ldc)
{
  const int tid  = threadIdx.x;
  const int lane = tid & 31;
  const int wave = tid >> 5;
  const int half = lane >> 4;
  const int l16  = lane & 15;
  const int mg   = wave & 3;              // 4 row-groups of 32
  const int ng   = wave >> 2;             // 2 col-groups of 32
  const int mbase = blockIdx.x * 128 + mg * 32;
  const int nbase = blockIdx.y * 64  + ng * 32;

  // Fragment feed pointers (ISA 16-bit 16x32 layout: lane l16 = row,
  // half selects K ranges [half*8, +8) and [16+half*8, +8)).
  const bf16* a0p = A  + (long)(mbase + l16) * Kpad + half * 8;
  const bf16* a1p = a0p + (long)16 * Kpad;
  const bf16* b0p = Wt + (long)(nbase + l16) * Kpad + half * 8;
  const bf16* b1p = b0p + (long)16 * Kpad;

  v8f acc00 = {}, acc01 = {}, acc10 = {}, acc11 = {};
  union U { uint4 q[2]; v16bf v; };

  for (int k0 = 0; k0 < Kpad; k0 += 32) {
    U a0, a1, b0, b1;
    a0.q[0] = *(const uint4*)(a0p + k0);
    a0.q[1] = *(const uint4*)(a0p + k0 + 16);
    a1.q[0] = *(const uint4*)(a1p + k0);
    a1.q[1] = *(const uint4*)(a1p + k0 + 16);
    b0.q[0] = *(const uint4*)(b0p + k0);
    b0.q[1] = *(const uint4*)(b0p + k0 + 16);
    b1.q[0] = *(const uint4*)(b1p + k0);
    b1.q[1] = *(const uint4*)(b1p + k0 + 16);
    acc00 = __builtin_amdgcn_wmma_f32_16x16x32_bf16(false, a0.v, false, b0.v, (short)0, acc00, false, false);
    acc01 = __builtin_amdgcn_wmma_f32_16x16x32_bf16(false, a0.v, false, b1.v, (short)0, acc01, false, false);
    acc10 = __builtin_amdgcn_wmma_f32_16x16x32_bf16(false, a1.v, false, b0.v, (short)0, acc10, false, false);
    acc11 = __builtin_amdgcn_wmma_f32_16x16x32_bf16(false, a1.v, false, b1.v, (short)0, acc11, false, false);
  }

  // Epilogue per C/D layout: lane -> col nt*16+l16, rows half*8+i.
  v8f accs[4]; accs[0] = acc00; accs[1] = acc01; accs[2] = acc10; accs[3] = acc11;
  #pragma unroll
  for (int t = 0; t < 4; ++t) {
    const int mt = t >> 1, nt = t & 1;
    const int n = nbase + nt * 16 + l16;
    if (n >= N) continue;
    const float bvv = bias ? bias[n] : 0.f;
    #pragma unroll
    for (int i = 0; i < 8; ++i) {
      const int m = mbase + mt * 16 + half * 8 + i;
      if (m >= M) continue;
      float v = accs[t][i] * scale + bvv;
      if (relu) v = fmaxf(v, 0.f);
      if (rowmask && !rowmask[m]) v = 0.f;
      C[(long)m * ldc + n] = v;
    }
  }
}

// ---------------------------------------------------------------------------
// tf0[c] = cos(time_b[c])  (Time2Vec of t=0; broadcast row for q inputs)
// ---------------------------------------------------------------------------
__global__ void tf0_kernel(const float* __restrict__ time_b, float* __restrict__ tf0)
{
  const int i = threadIdx.x;
  if (i < TD_) tf0[i] = __cosf(time_b[i]);
}

// ---------------------------------------------------------------------------
// Attention core (reassociated): per node r, read wq_eff[r] (2 x 356),
// build each kv row once (gathered nbr feat | edge feat | Time2Vec(dt)),
// cache it in LDS, compute masked-softmax weights, then overwrite
// wq_eff[r] with ctx[r] = sum_k w[h,k] * kv[k,:].  One wave per node.
// ---------------------------------------------------------------------------
__global__ __launch_bounds__(128) void attn_core(
    int n, float* __restrict__ wqctx, int* __restrict__ validany,
    const int* __restrict__ nids, const float* __restrict__ ntimes,
    const float* __restrict__ tarr, const float* __restrict__ efeat,
    const float* __restrict__ nbr, int nbr_gather, int nbr_ld,
    const float* __restrict__ time_w, const float* __restrict__ time_b)
{
  __shared__ float kvbuf[4][K_NBR][360];   // 57.6 KB
  const int lane = threadIdx.x & 31;
  const int slot = threadIdx.x >> 5;
  const int r = blockIdx.x * 4 + slot;
  if (r >= n) return;

  const float tq = tarr[r];
  float* wq = wqctx + (size_t)r * (2 * KD_);

  float sc0[K_NBR], sc1[K_NBR];
  int anyv = 0;
  for (int k = 0; k < K_NBR; ++k) {
    const int nid   = nids[r*K_NBR + k];
    const int valid = nid > 0;
    anyv |= valid;
    const float dt = tq - ntimes[r*K_NBR + k];
    const size_t nroff = (size_t)(nbr_gather ? nid : (r*K_NBR + k)) * nbr_ld;
    const size_t eoff  = (size_t)(r*K_NBR + k) * ED_;
    float p0 = 0.f, p1 = 0.f;
    for (int c = lane; c < KD_; c += 32) {
      float v;
      if (c < ND_)            v = nbr[nroff + c];
      else if (c < ND_ + ED_) v = efeat[eoff + (c - ND_)];
      else { const int tc = c - ND_ - ED_; v = __cosf(fmaf(dt, time_w[tc], time_b[tc])); }
      kvbuf[slot][k][c] = v;
      p0 = fmaf(v, wq[c],       p0);
      p1 = fmaf(v, wq[KD_ + c], p1);
    }
    // butterfly reduce -> all lanes get the full dot product
    for (int off = 16; off > 0; off >>= 1) {
      p0 += __shfl_xor(p0, off, 32);
      p1 += __shfl_xor(p1, off, 32);
    }
    sc0[k] = valid ? p0 : -1e9f;   // 1/sqrt(DH) already folded into wq_eff
    sc1[k] = valid ? p1 : -1e9f;
  }

  // softmax (redundant per lane, registers only)
  float m0 = -1e30f, m1 = -1e30f;
  for (int k = 0; k < K_NBR; ++k) { m0 = fmaxf(m0, sc0[k]); m1 = fmaxf(m1, sc1[k]); }
  float w0[K_NBR], w1[K_NBR], z0 = 0.f, z1 = 0.f;
  for (int k = 0; k < K_NBR; ++k) {
    w0[k] = __expf(sc0[k] - m0); z0 += w0[k];
    w1[k] = __expf(sc1[k] - m1); z1 += w1[k];
  }
  const float i0 = 1.f / z0, i1 = 1.f / z1;
  for (int k = 0; k < K_NBR; ++k) { w0[k] *= i0; w1[k] *= i1; }

  // context aggregation, in place over wq_eff (all reads of wq done above)
  for (int c = lane; c < KD_; c += 32) {
    float a0 = 0.f, a1 = 0.f;
    #pragma unroll
    for (int k = 0; k < K_NBR; ++k) {
      const float v = kvbuf[slot][k][c];
      a0 = fmaf(w0[k], v, a0);
      a1 = fmaf(w1[k], v, a1);
    }
    wq[c]       = a0;
    wq[KD_ + c] = a1;
  }
  if (lane == 0) validany[r] = anyv;
}

// ---------------------------------------------------------------------------
// Host-side orchestration
// ---------------------------------------------------------------------------
struct WeightsL {
  const float *Wq, *bq, *Wk, *Wv, *bv, *Wo, *bo, *M1w, *M1b, *M2w, *M2b;
};

static void run_gemm(hipStream_t st, Seg s0, Seg s1, int M, int Kt, int N,
                     const float* W, int ldw, int wtrans, const float* bias,
                     float scale, int relu, const int* rowmask,
                     float* C, int ldc, bf16* Ap, bf16* Wp)
{
  const int Kpad = (Kt + 31) & ~31;
  const long Mpad = ((long)M + 127) & ~127L;
  const int Npad = (N + 63) & ~63;

  const long atotal = Mpad * (Kpad >> 3);
  pack_a_kernel<<<dim3((unsigned)((atotal + 255) / 256)), dim3(256), 0, st>>>(
      s0, s1, M, Kt, Kpad, Mpad, Ap);
  const int wtotal = Npad * (Kpad >> 3);
  pack_w_kernel<<<dim3((wtotal + 255) / 256), dim3(256), 0, st>>>(
      W, ldw, wtrans, Kt, N, Kpad, Npad, Wp);
  gemm_wmma_packed<<<dim3((unsigned)(Mpad / 128), Npad / 64), dim3(256), 0, st>>>(
      Ap, Wp, M, N, Kpad, bias, scale, relu, rowmask, C, ldc);
}

static void emb1_run(hipStream_t st, int n,
                     Seg qnode, Seg mnode,
                     const int* nids, const float* ntimes, const float* tarr,
                     const float* efeat, const float* nbr, int nbr_gather, int nbr_ld,
                     const float* time_w, const float* time_b, const WeightsL& w,
                     float* TF0, int* VALID, float* Q, float* WQ, float* OUT2,
                     float* H1, bf16* Ap, bf16* Wp, float* dest, int dest_ld)
{
  const float invsq = 0.09365858f;   // 1/sqrt(114), folded into wq_eff
  // q = [node_feat || tf0] @ Wq + bq
  run_gemm(st, qnode, segB(TF0, TD_), n, QD_, QD_, w.Wq, QD_, 0, w.bq,
           1.f, 0, nullptr, Q, QD_, Ap, Wp);
  // wq_eff_h = (q_h @ Wk_h^T) / sqrt(DH)
  for (int h = 0; h < 2; ++h)
    run_gemm(st, segD(Q + h*DH_, DH_, QD_), segN(), n, DH_, KD_,
             w.Wk + h*DH_, QD_, 1, nullptr, invsq, 0, nullptr,
             WQ + h*KD_, 2*KD_, Ap, Wp);
  // scores -> softmax -> ctx (in place over WQ)
  attn_core<<<dim3((n + 3) / 4), dim3(128), 0, st>>>(
      n, WQ, VALID, nids, ntimes, tarr, efeat, nbr, nbr_gather, nbr_ld,
      time_w, time_b);
  // out_h = ctx_h @ Wv_h + bv_h   (softmax weights sum to 1 -> bias exact)
  for (int h = 0; h < 2; ++h)
    run_gemm(st, segD(WQ + h*KD_, KD_, 2*KD_), segN(), n, KD_, DH_,
             w.Wv + h*DH_, QD_, 0, w.bv + h*DH_, 1.f, 0, nullptr,
             Q + h*DH_, QD_, Ap, Wp);
  // out2 = (out @ Wo + bo) * validany
  run_gemm(st, segD(Q, QD_, QD_), segN(), n, QD_, QD_, w.Wo, QD_, 0, w.bo,
           1.f, 0, VALID, OUT2, QD_, Ap, Wp);
  // h1 = relu([out2 || node_feat] @ M1w + M1b)
  run_gemm(st, segD(OUT2, QD_, QD_), mnode, n, QD_ + ND_, EMB_, w.M1w, EMB_, 0,
           w.M1b, 1.f, 1, nullptr, H1, EMB_, Ap, Wp);
  // emb = h1 @ M2w + M2b
  run_gemm(st, segD(H1, EMB_, EMB_), segN(), n, EMB_, EMB_, w.M2w, EMB_, 0,
           w.M2b, 1.f, 0, nullptr, dest, dest_ld, Ap, Wp);
}

extern "C" void kernel_launch(void* const* d_in, const int* in_sizes, int n_in,
                              void* d_out, int out_size, void* d_ws, size_t ws_size,
                              hipStream_t stream)
{
  (void)in_sizes; (void)n_in; (void)out_size; (void)ws_size;

  const float* staticf = (const float*)d_in[0];
  const int*   src_ids = (const int*)  d_in[1];
  const int*   dst_ids = (const int*)  d_in[2];
  const float* timeq   = (const float*)d_in[3];
  const float* time_w  = (const float*)d_in[16];
  const float* time_b  = (const float*)d_in[17];
  const float* Wq  = (const float*)d_in[18];
  const float* bq  = (const float*)d_in[19];
  const float* Wk  = (const float*)d_in[20];
  // d_in[21] (bk): constant per-(n,h) shift of all k scores -> softmax-invariant
  const float* Wv  = (const float*)d_in[22];
  const float* bv  = (const float*)d_in[23];
  const float* Wo  = (const float*)d_in[24];
  const float* bo  = (const float*)d_in[25];
  const float* M1w = (const float*)d_in[26];
  const float* M1b = (const float*)d_in[27];
  const float* M2w = (const float*)d_in[28];
  const float* M2b = (const float*)d_in[29];
  float* out = (float*)d_out;

  // ---- workspace layout (floats, 64-element aligned blocks) ------------
  float* f = (float*)d_ws;
  size_t off = 0;
  auto takef = [&](size_t cnt) -> float* {
    float* p = f + off; off += (cnt + 63) & ~(size_t)63; return p;
  };
  const size_t NMAX = (size_t)B_SZ * K_NBR;           // 40960
  float* TF0   = takef(128);
  int*   VALID = (int*)takef(NMAX);
  float* Q     = takef(NMAX * QD_);                   // q / attn out
  float* WQ    = takef(NMAX * 2 * KD_);               // wq_eff, then ctx
  float* OUT2  = takef(NMAX * QD_);
  float* H1    = takef(NMAX * EMB_);
  float* EMBN  = takef(NMAX * EMB_);                  // hop-1 neighbor embeddings
  float* EMB0  = takef((size_t)B_SZ * EMB_);          // node embedding (level 0)
  bf16*  Ap    = (bf16*)takef(NMAX * 384 / 2);        // packed A (bf16)
  bf16*  Wp    = (bf16*)takef((size_t)384 * 384 / 2); // packed W (bf16)

  auto slice = [&](int l) -> WeightsL {
    WeightsL w;
    w.Wq  = Wq  + (size_t)l * QD_ * QD_;   w.bq  = bq  + (size_t)l * QD_;
    w.Wk  = Wk  + (size_t)l * KD_ * QD_;
    w.Wv  = Wv  + (size_t)l * KD_ * QD_;   w.bv  = bv  + (size_t)l * QD_;
    w.Wo  = Wo  + (size_t)l * QD_ * QD_;   w.bo  = bo  + (size_t)l * QD_;
    w.M1w = M1w + (size_t)l * (QD_ + ND_) * EMB_;  w.M1b = M1b + (size_t)l * EMB_;
    w.M2w = M2w + (size_t)l * EMB_ * EMB_;         w.M2b = M2b + (size_t)l * EMB_;
    return w;
  };
  const WeightsL w0 = slice(0), w1 = slice(1);

  tf0_kernel<<<1, 128, 0, stream>>>(time_b, TF0);

  for (int role = 0; role < 2; ++role) {
    const int*   ids    = role ? dst_ids : src_ids;
    const int    bi     = 4 + role * 6;
    const int*   n1_ids = (const int*)  d_in[bi + 0];
    const float* n1_t   = (const float*)d_in[bi + 1];
    const float* n1_e   = (const float*)d_in[bi + 2];
    const int*   n0_ids = (const int*)  d_in[bi + 3];
    const float* n0_t   = (const float*)d_in[bi + 4];
    const float* n0_e   = (const float*)d_in[bi + 5];

    // level-0 embedding of the B query nodes
    emb1_run(stream, B_SZ,
             segG(staticf, ids, ND_, ND_), segG(staticf, ids, ND_, ND_),
             n1_ids, n1_t, timeq, n1_e, staticf, /*gather=*/1, ND_,
             time_w, time_b, w0, TF0, VALID, Q, WQ, OUT2, H1, Ap, Wp,
             EMB0, EMB_);
    // level-0 embedding of the B*K hop-1 neighbors
    emb1_run(stream, B_SZ * K_NBR,
             segG(staticf, n1_ids, ND_, ND_), segG(staticf, n1_ids, ND_, ND_),
             n0_ids, n0_t, n1_t, n0_e, staticf, /*gather=*/1, ND_,
             time_w, time_b, w0, TF0, VALID, Q, WQ, OUT2, H1, Ap, Wp,
             EMBN, EMB_);
    // level-1: neighbor features are the computed embeddings (direct rows)
    emb1_run(stream, B_SZ,
             segD(EMB0, EMB_, EMB_), segG(staticf, ids, ND_, ND_),
             n1_ids, n1_t, timeq, n1_e, EMBN, /*gather=*/0, EMB_,
             time_w, time_b, w1, TF0, VALID, Q, WQ, OUT2, H1, Ap, Wp,
             out + (size_t)role * B_SZ * EMB_, EMB_);
  }
}
